// WindowAttention_12987981103413
// MI455X (gfx1250) — matrix operand order repaired and checked by reading, not compiled, not run on hardware
//
#include <hip/hip_runtime.h>
#include <math.h>

typedef __attribute__((ext_vector_type(16))) _Float16 v16h;
typedef __attribute__((ext_vector_type(8)))  float    v8f;

#define NWIN  256
#define VTOK  343
#define VPAD  352
#define CDIM  96
#define HEADS 3
#define HD    32
#define TABLE 2197
#define QSCALE 0.17677669529663689f   /* 32^-0.5 */

union F16Frag { v16h v; _Float16 h[16]; unsigned u[8]; };
union F32Acc  { v8f  v; float    f[8]; };
union HPack   { unsigned u; _Float16 h[2]; };

static __device__ __forceinline__ v8f wmma_f16(v16h a, v16h b, v8f c) {
  return __builtin_amdgcn_wmma_f32_16x16x32_f16(false, a, false, b, (short)0, c,
                                                false, false);
}

// -------------------------------------------------------------------------
// Stage 1: qkv = x @ w_qkv^T   (87808 x 96) @ (96 x 288)
// 8 waves/block, each wave = 16 rows. Writes f16 Q (scaled), K, V^T tiles.
// n-tile loop fully unrolled so which/head/d-base are compile-time constants.
// -------------------------------------------------------------------------
__global__ __launch_bounds__(256) void qkv_proj_kernel(
    const float* __restrict__ x, const float* __restrict__ w,
    _Float16* __restrict__ Qh, _Float16* __restrict__ Kh,
    _Float16* __restrict__ Vh)
{
  extern __shared__ _Float16 smem[];          // w as f16: [288][98] (odd-dword row)
  const int tid = threadIdx.x;
  // stage w into LDS as packed f16 pairs (dword stores)
  for (int i = tid; i < 288 * 48; i += 256) {
    const int r = i / 48, cd = i - r * 48;
    HPack p;
    p.h[0] = (_Float16)w[r * 96 + 2 * cd];
    p.h[1] = (_Float16)w[r * 96 + 2 * cd + 1];
    ((unsigned*)smem)[r * 49 + cd] = p.u;
  }
  __syncthreads();

  const int wave = tid >> 5, lane = tid & 31;
  const int ln = lane & 15, hl = lane >> 4;
  const int rowBase = blockIdx.x * 128 + wave * 16;
  const int arow = rowBase + ln;              // always < 87808 (686*128 exact)

  // A fragments: rows of x, K = 16*hl + j, per 32-wide k-step
  F16Frag a[3];
  #pragma unroll
  for (int ks = 0; ks < 3; ++ks) {
    #pragma unroll
    for (int jb = 0; jb < 2; ++jb) {
      const float* px = x + (size_t)arow * 96 + 32 * ks + 16 * hl + 8 * jb;
      #pragma unroll
      for (int i = 0; i < 8; ++i) a[ks].h[jb * 8 + i] = (_Float16)px[i];
    }
  }

  // hoisted per-row output bases (C layout row M = r + 8*hl)
  size_t qkBase[8], vBase[8];
  #pragma unroll
  for (int r = 0; r < 8; ++r) {
    const int gr = rowBase + r + 8 * hl;
    const int n  = gr / 343;
    const int vv = gr - n * 343;
    qkBase[r] = ((size_t)n * 3 * VPAD + vv) * HD;       // + hh*VPAD*HD + d
    vBase[r]  = (size_t)n * 3 * HD * VPAD + vv;         // + hh*HD*VPAD + d*VPAD
  }

  #pragma unroll
  for (int nt = 0; nt < 18; ++nt) {
    v8f acc = {};
    #pragma unroll
    for (int ks = 0; ks < 3; ++ks) {
      // B = w^T : B[k][col] = w[col][k]; lane half j -> k = 32ks + 16*hl + j
      F16Frag b;
      const unsigned* pw =
          (const unsigned*)(smem + (nt * 16 + ln) * 98 + 32 * ks + 16 * hl);
      #pragma unroll
      for (int m = 0; m < 8; ++m) b.u[m] = pw[m];
      acc = wmma_f16(a[ks].v, b.v, acc);
    }
    F32Acc o; o.v = acc;
    // 16-wide tile never straddles a 96- or 32-column boundary:
    const int colBase = nt * 16;              // compile-time constant
    const int which = colBase / 96;           // 0=q 1=k 2=v (constant)
    const int cc0 = colBase - which * 96;
    const int hh = cc0 >> 5;                  // head (constant)
    const int d = (cc0 & 31) + ln;            // dim 0..31
    #pragma unroll
    for (int r = 0; r < 8; ++r) {
      const float val = o.f[r];
      if (which == 0)
        Qh[qkBase[r] + (size_t)hh * (VPAD * HD) + d] = (_Float16)(val * QSCALE);
      else if (which == 1)
        Kh[qkBase[r] + (size_t)hh * (VPAD * HD) + d] = (_Float16)val;
      else
        Vh[vBase[r] + (size_t)hh * (HD * VPAD) + (size_t)d * VPAD] = (_Float16)val;
    }
  }
}

// -------------------------------------------------------------------------
// Stage 2: per (n, h, 64-row query block): S = QK^T + bias + mask,
// exact softmax (full row in registers), O = softmax(S) @ V.
// 4 waves/block, each wave = 16 query rows.
// -------------------------------------------------------------------------
__global__ __launch_bounds__(128) void attn_kernel(
    const _Float16* __restrict__ Qh, const _Float16* __restrict__ Kh,
    const _Float16* __restrict__ Vh,
    const float* __restrict__ mask, const float* __restrict__ bias,
    const int* __restrict__ rel, float* __restrict__ out)
{
  extern __shared__ _Float16 smem[];
  _Float16* Kt  = smem;                        // [352][34]  (K rows, dims)
  _Float16* VtT = smem + 352 * 34;             // [32][354]  (dims, keys)
  _Float16* Pb  = VtT + 32 * 354;              // [4][16][354] per-wave probs

  const int mb = blockIdx.x;                   // 0..5
  const int h  = blockIdx.y;                   // 0..2
  const int n  = blockIdx.z;                   // 0..255
  const int tid = threadIdx.x;
  const int wave = tid >> 5, lane = tid & 31;
  const int ln = lane & 15, hl = lane >> 4;
  const size_t nh = (size_t)(n * 3 + h);
  const _Float16* Qg = Qh + nh * (size_t)(VPAD * HD);
  const _Float16* Kg = Kh + nh * (size_t)(VPAD * HD);
  const _Float16* Vg = Vh + nh * (size_t)(HD * VPAD);

  // K tile -> LDS (zero-pad keys >= 343)
  for (int i = tid; i < 352 * 16; i += 128) {  // dword granularity
    const int key = i >> 4, off = i & 15;
    unsigned val = 0u;
    if (key < VTOK) val = ((const unsigned*)Kg)[key * 16 + off];
    ((unsigned*)Kt)[key * 17 + off] = val;
  }
  // V^T tile -> LDS (zero-pad keys >= 343)
  for (int i = tid; i < 32 * 176; i += 128) {
    const int dim = i / 176, ko = i - dim * 176;
    const int k0 = 2 * ko;
    HPack p;
    p.h[0] = (k0     < VTOK) ? Vg[dim * VPAD + k0]     : (_Float16)0.0f;
    p.h[1] = (k0 + 1 < VTOK) ? Vg[dim * VPAD + k0 + 1] : (_Float16)0.0f;
    ((unsigned*)VtT)[dim * 177 + ko] = p.u;
  }
  __syncthreads();

  const int qbase = mb * 64 + wave * 16;

  // Q fragment (16 rows x 32 dims), zero for padded rows
  F16Frag aq;
  {
    const int row = qbase + ln;
    if (row < VTOK) {
      #pragma unroll
      for (int jb = 0; jb < 2; ++jb) {
        const unsigned* pq = (const unsigned*)(Qg + row * HD + 16 * hl + 8 * jb);
        #pragma unroll
        for (int m = 0; m < 4; ++m) aq.u[jb * 4 + m] = pq[m];
      }
    } else {
      #pragma unroll
      for (int m = 0; m < 8; ++m) aq.u[m] = 0u;
    }
  }

  // ---- S = QK^T + bias + mask, full 16x352 row block in registers ----
  float sacc[22][8];
  #pragma unroll
  for (int t = 0; t < 22; ++t) {
    F16Frag bk;  // B[k=dim][n=key] = Kt[key][dim]
    const unsigned* pk = (const unsigned*)(Kt + (t * 16 + ln) * 34 + 16 * hl);
    #pragma unroll
    for (int m = 0; m < 8; ++m) bk.u[m] = pk[m];
    F32Acc s; s.v = wmma_f16(aq.v, bk.v, (v8f){});
    const int col = t * 16 + ln;
    #pragma unroll
    for (int r = 0; r < 8; ++r) {
      const int qr = qbase + r + 8 * hl;
      float pval = -1e30f;
      if (qr < VTOK && col < VTOK) {
        pval = s.f[r] + bias[h * TABLE + rel[qr * VTOK + col]]
             + mask[((size_t)n * VTOK + qr) * VTOK + col];
      }
      sacc[t][r] = pval;
    }
  }

  // ---- exact softmax: row max / exp / row sum; probs -> LDS as f16 ----
  _Float16* Pw = Pb + wave * 16 * 354;
  float linv[8];
  #pragma unroll
  for (int r = 0; r < 8; ++r) {
    float m = -1e30f;
    #pragma unroll
    for (int t = 0; t < 22; ++t) m = fmaxf(m, sacc[t][r]);
    #pragma unroll
    for (int off = 1; off < 16; off <<= 1) m = fmaxf(m, __shfl_xor(m, off, 32));
    const int qlocal = r + 8 * hl;
    float sum = 0.f;
    #pragma unroll
    for (int t = 0; t < 22; ++t) {
      const float e = __expf(sacc[t][r] - m);
      Pw[qlocal * 354 + t * 16 + ln] = (_Float16)e;
      sum += e;
    }
    #pragma unroll
    for (int off = 1; off < 16; off <<= 1) sum += __shfl_xor(sum, off, 32);
    linv[r] = __builtin_amdgcn_rcpf(sum);    // ~1ulp reciprocal, v_rcp_f32
  }
  // Pw is wave-private; in-wave DS ordering handled by compiler-inserted waits.

  // ---- O = P @ V : 2 n-halves x 11 k-steps of 32 keys ----
  #pragma unroll
  for (int nh2 = 0; nh2 < 2; ++nh2) {
    v8f o = {};
    #pragma unroll
    for (int ks = 0; ks < 11; ++ks) {
      F16Frag ap, bv;
      #pragma unroll
      for (int jb = 0; jb < 2; ++jb) {
        const unsigned* pp =
            (const unsigned*)(Pw + ln * 354 + 32 * ks + 16 * hl + 8 * jb);
        #pragma unroll
        for (int m = 0; m < 4; ++m) ap.u[jb * 4 + m] = pp[m];
      }
      const unsigned* pv =
          (const unsigned*)(VtT + (nh2 * 16 + ln) * 354 + 32 * ks + 16 * hl);
      #pragma unroll
      for (int m = 0; m < 8; ++m) bv.u[m] = pv[m];
      o = wmma_f16(ap.v, bv.v, o);
    }
    F32Acc oo; oo.v = o;
    #pragma unroll
    for (int r = 0; r < 8; ++r) {
      const int qr = qbase + r + 8 * hl;
      if (qr < VTOK) {
        out[((size_t)n * VTOK + qr) * CDIM + h * HD + nh2 * 16 + ln] =
            oo.f[r] * linv[r];
      }
    }
  }
}

// -------------------------------------------------------------------------
extern "C" void kernel_launch(void* const* d_in, const int* in_sizes, int n_in,
                              void* d_out, int out_size, void* d_ws, size_t ws_size,
                              hipStream_t stream) {
  (void)in_sizes; (void)n_in; (void)out_size; (void)ws_size;
  const float* x    = (const float*)d_in[0];
  const float* mask = (const float*)d_in[1];
  const float* w    = (const float*)d_in[2];
  const float* bias = (const float*)d_in[3];
  const int*   rel  = (const int*)d_in[4];
  float* out = (float*)d_out;

  const size_t per = (size_t)NWIN * HEADS * VPAD * HD;  // 8,650,752 halves
  _Float16* Qh = (_Float16*)d_ws;
  _Float16* Kh = Qh + per;
  _Float16* Vh = Kh + per;   // total ws use: 3*per*2 B ~= 49.5 MB

  const size_t sh1 = (size_t)288 * 98 * sizeof(_Float16);          // 56,448 B
  qkv_proj_kernel<<<dim3(686), dim3(256), sh1, stream>>>(x, w, Qh, Kh, Vh);

  const size_t sh2 =
      (size_t)(352 * 34 + 32 * 354 + 4 * 16 * 354) * sizeof(_Float16); // 91,904 B
  attn_kernel<<<dim3(6, 3, 256), dim3(128), sh2, stream>>>(
      Qh, Kh, Vh, mask, bias, rel, out);
}
